// MusicTransformer_87720412054210
// MI455X (gfx1250) — matrix operand order repaired and checked by reading, not compiled, not run on hardware
//
#include <hip/hip_runtime.h>

// ---------------- model constants ----------------
constexpr int Vv = 1024, Dv = 768, Hh = 12, HDv = 64, Lv = 8, Fv = 3072, Tv = 1024, Bb = 2;
constexpr int Mrows = Bb * Tv;           // 2048 token rows
constexpr float EPSv = 1e-5f;

// ---------------- vector types ----------------
typedef _Float16 v16h __attribute__((ext_vector_type(16)));
typedef _Float16 v4h  __attribute__((ext_vector_type(4)));
typedef float    v8f  __attribute__((ext_vector_type(8)));
typedef unsigned int v8u __attribute__((ext_vector_type(8)));

// ---------------- utility kernels ----------------
__global__ __launch_bounds__(256) void cvt_f32_f16_kernel(const float* __restrict__ in,
                                                          _Float16* __restrict__ out, long n) {
  long i = (long)blockIdx.x * 256 + threadIdx.x;
  if (i < n) out[i] = (_Float16)in[i];
}

// in: [L][K][N] f32 row-major  ->  out: [L][N][K] f16 row-major
__global__ __launch_bounds__(256) void cvt_transpose_kernel(const float* __restrict__ in,
                                                            _Float16* __restrict__ out,
                                                            int Kd, int Nd) {
  int i2 = blockIdx.x * 256 + threadIdx.x;      // index in one layer's N*K output
  int l  = blockIdx.y;
  long kn = (long)Kd * Nd;
  if (i2 < kn) {
    int n = i2 / Kd;
    int k = i2 - n * Kd;
    out[l * kn + i2] = (_Float16)in[l * kn + (long)k * Nd + n];
  }
}

__global__ __launch_bounds__(256) void embed_kernel(const int* __restrict__ ids,
                                                    const float* __restrict__ tok,
                                                    const float* __restrict__ pos,
                                                    float* __restrict__ x) {
  int row = blockIdx.x;                 // 0..2047
  int t = row & (Tv - 1);
  int id = ids[row];
  const float* tr = tok + (long)id * Dv;
  const float* pr = pos + (long)t * Dv;
  float* xr = x + (long)row * Dv;
  for (int j = threadIdx.x; j < Dv; j += 256) xr[j] = tr[j] + pr[j];
}

__device__ inline float block_reduce_sum(float v, float* red) {
  int tid = threadIdx.x;
  red[tid] = v; __syncthreads();
  for (int s = 128; s > 0; s >>= 1) { if (tid < s) red[tid] += red[tid + s]; __syncthreads(); }
  float r = red[0]; __syncthreads(); return r;
}
__device__ inline float block_reduce_max(float v, float* red) {
  int tid = threadIdx.x;
  red[tid] = v; __syncthreads();
  for (int s = 128; s > 0; s >>= 1) { if (tid < s) red[tid] = fmaxf(red[tid], red[tid + s]); __syncthreads(); }
  float r = red[0]; __syncthreads(); return r;
}

__global__ __launch_bounds__(256) void layernorm_f16_kernel(const float* __restrict__ x,
                                                            const float* __restrict__ w,
                                                            const float* __restrict__ b,
                                                            _Float16* __restrict__ out) {
  __shared__ float red[256];
  long row = blockIdx.x;
  const float* xr = x + row * Dv;
  float s = 0.f;
  for (int j = threadIdx.x; j < Dv; j += 256) s += xr[j];
  float mu = block_reduce_sum(s, red) * (1.0f / Dv);
  float vs = 0.f;
  for (int j = threadIdx.x; j < Dv; j += 256) { float d = xr[j] - mu; vs += d * d; }
  float var = block_reduce_sum(vs, red) * (1.0f / Dv);
  float rs = rsqrtf(var + EPSv);
  _Float16* orow = out + row * Dv;
  for (int j = threadIdx.x; j < Dv; j += 256)
    orow[j] = (_Float16)((xr[j] - mu) * rs * w[j] + b[j]);
}

// causal softmax, in place on f16 scores; applies 1/sqrt(HD) scale
__global__ __launch_bounds__(256) void softmax_kernel(_Float16* __restrict__ att, float scale) {
  __shared__ float red[256];
  int r = blockIdx.x;                       // query row 0..T-1
  long slice = blockIdx.y;                  // b*H + h
  _Float16* row = att + slice * (long)Tv * Tv + (long)r * Tv;
  float m = -1e30f;
  for (int j = threadIdx.x; j <= r; j += 256) m = fmaxf(m, (float)row[j] * scale);
  m = block_reduce_max(m, red);
  float s = 0.f;
  for (int j = threadIdx.x; j <= r; j += 256) s += __expf((float)row[j] * scale - m);
  s = block_reduce_sum(s, red);
  float inv = 1.0f / s;
  for (int j = threadIdx.x; j < Tv; j += 256) {
    float v = (j <= r) ? __expf((float)row[j] * scale - m) * inv : 0.0f;
    row[j] = (_Float16)v;
  }
}

// ---------------- WMMA GEMM with async double-buffered LDS staging ----------------
// C[M,N] = epilogue( A[M,K](f16,lda) * B(f16) [+bias] [gelu] ) [+resid]
// BT=false: B is [K,N] row-major (ldb)  -- synchronous VGPR transpose staging
// BT=true : B is [N,K] row-major (ldb)  -- async GLOBAL_LOAD_ASYNC_TO_LDS_B128 staging
// Batched via grid.z: z -> (zb=z/Hdim, zh=z%Hdim) with independent strides.
constexpr int BMt = 128, BNt = 64, BKt = 32;
constexpr int LSA = 40;   // LDS row stride in halves: 80B rows keep 16B async alignment,
                          // 20-dword stride -> no bank conflicts across the 16 M/N lanes

#define ASYNC_CP_B128(ldsOff, gaddr) \
  asm volatile("global_load_async_to_lds_b128 %0, %1, off" :: "v"(ldsOff), "v"(gaddr) : "memory")
#define WAIT_ASYNC0() asm volatile("s_wait_asynccnt 0x0" ::: "memory")

template<bool BT, bool OUT_LP, bool GELU_ACT, bool HAS_BIAS, bool HAS_RES>
__global__ __launch_bounds__(256)
void gemm_wmma(const _Float16* __restrict__ A, const _Float16* __restrict__ Bm,
               float* __restrict__ Cf, _Float16* __restrict__ Ch,
               const float* __restrict__ bias, const float* __restrict__ resid,
               int M, int N, int K, int lda, int ldb, int ldc, int Hdim,
               long sAb, long sAh, long sBb, long sBh, long sCb, long sCh) {
  __shared__ _Float16 As[2][BMt][LSA];
  __shared__ _Float16 Bs[2][BNt][LSA];   // always [n][k]

  const int z  = blockIdx.z;
  const int zb = z / Hdim, zh = z % Hdim;
  A  += (long)zb * sAb + (long)zh * sAh;
  Bm += (long)zb * sBb + (long)zh * sBh;
  const long coff = (long)zb * sCb + (long)zh * sCh;

  const int rowBase = blockIdx.y * BMt;
  const int colBase = blockIdx.x * BNt;
  const int tid  = threadIdx.x;
  const int lane = tid & 31;
  const int wave = tid >> 5;
  const int wm = (wave & 3) * 32;     // wave tile origin (M)
  const int wn = (wave >> 2) * 32;    // wave tile origin (N)
  const int mlo   = lane & 15;
  const int khalf = (lane >> 4) << 3; // A: k offset 0/8 by lane half
  const int ksB   = (lane >> 4) << 3; // B: k offset 0/8 by lane half

  // ---- per-thread async chunk assignment (16B chunks) ----
  const int ar = tid >> 2;                  // 0..63  (and +64 for 2nd A chunk)
  const int ac = (tid & 3) * 8;             // half offset in row: 0,8,16,24
  unsigned long long gA0 = (unsigned long long)(A + (long)(rowBase + ar) * lda + ac);
  unsigned long long gA1 = gA0 + (unsigned long long)((long)64 * lda * 2);
  unsigned long long gB  = 0;
  if constexpr (BT)
    gB = (unsigned long long)(Bm + (long)(colBase + ar) * ldb + ac);

  unsigned aoff0[2], aoff1[2], boff[2];
  #pragma unroll
  for (int bb = 0; bb < 2; ++bb) {
    aoff0[bb] = (unsigned)(unsigned long long)&As[bb][ar][ac];
    aoff1[bb] = (unsigned)(unsigned long long)&As[bb][ar + 64][ac];
    boff[bb]  = (unsigned)(unsigned long long)&Bs[bb][ar][ac];
  }

  auto stage = [&](int kk, int buf) {
    unsigned long long kb2 = (unsigned long long)kk * 2;   // byte advance along K
    ASYNC_CP_B128(aoff0[buf], gA0 + kb2);
    ASYNC_CP_B128(aoff1[buf], gA1 + kb2);
    if constexpr (BT) {
      ASYNC_CP_B128(boff[buf], gB + kb2);
    } else {
      #pragma unroll
      for (int it = 0; it < 2; ++it) {
        int idx = tid + it * 256;
        int kr = idx >> 4;
        int n = (idx & 15) << 2;
        v4h vv = *(const v4h*)(Bm + (long)(kk + kr) * ldb + (colBase + n));
        #pragma unroll
        for (int q = 0; q < 4; ++q) Bs[buf][n + q][kr] = vv[q];
      }
    }
  };

  v8f acc[2][2] = {};

  stage(0, 0);
  WAIT_ASYNC0();
  __syncthreads();

  const int nsteps = K / BKt;
  for (int i = 0; i < nsteps; ++i) {
    const int cur = i & 1;
    if (i + 1 < nsteps) stage((i + 1) * BKt, cur ^ 1);

    // ---- fragments per ISA 7.12.2 16-bit layouts ----
    v16h af[2], bfv[2];
    #pragma unroll
    for (int t = 0; t < 2; ++t) {
      int m = wm + t * 16 + mlo;
      v8u au;
      #pragma unroll
      for (int p = 0; p < 8; ++p) {
        int kb = ((p & 4) << 2) + ((p & 3) << 1) + khalf;   // K pair base
        au[p] = *(const unsigned int*)&As[cur][m][kb];
      }
      af[t] = __builtin_bit_cast(v16h, au);
      int n = wn + t * 16 + mlo;
      v8u bu;
      #pragma unroll
      for (int p = 0; p < 8; ++p) bu[p] = *(const unsigned int*)&Bs[cur][n][((p & 4) << 2) + ((p & 3) << 1) + ksB];
      bfv[t] = __builtin_bit_cast(v16h, bu);
    }
    #pragma unroll
    for (int mt = 0; mt < 2; ++mt)
      #pragma unroll
      for (int nt = 0; nt < 2; ++nt)
        acc[mt][nt] = __builtin_amdgcn_wmma_f32_16x16x32_f16(
            false, af[mt], false, bfv[nt], (short)0, acc[mt][nt], false, false);

    WAIT_ASYNC0();
    __syncthreads();
  }

  // ---- epilogue: bias -> gelu -> residual -> store (fully branch-free) ----
  const int rOff = (lane >> 4) << 3;  // C rows 0..7 / 8..15 by lane half
  #pragma unroll
  for (int mt = 0; mt < 2; ++mt) {
    #pragma unroll
    for (int nt = 0; nt < 2; ++nt) {
      int col = colBase + wn + nt * 16 + mlo;
      float bv = 0.0f;
      if constexpr (HAS_BIAS) bv = bias[col];
      #pragma unroll
      for (int r = 0; r < 8; ++r) {
        int row = rowBase + wm + mt * 16 + rOff + r;
        float vv = acc[mt][nt][r] + bv;
        if constexpr (GELU_ACT) vv = 0.5f * vv * (1.0f + erff(vv * 0.70710678118f));
        long o = (long)row * ldc + col;
        if constexpr (HAS_RES) vv += resid[o];
        if constexpr (OUT_LP) Ch[coff + o] = (_Float16)vv;
        else                  Cf[coff + o] = vv;
      }
    }
  }
}

// ---------------- host orchestration ----------------
extern "C" void kernel_launch(void* const* d_in, const int* in_sizes, int n_in,
                              void* d_out, int out_size, void* d_ws, size_t ws_size,
                              hipStream_t stream) {
  (void)in_sizes; (void)n_in; (void)out_size; (void)ws_size;
  const int*   ids  = (const int*)d_in[0];
  const float* tok  = (const float*)d_in[1];
  const float* pos  = (const float*)d_in[2];
  const float* ln1w = (const float*)d_in[3];
  const float* ln1b = (const float*)d_in[4];
  const float* qkvw = (const float*)d_in[5];
  const float* outw = (const float*)d_in[6];
  const float* ln2w = (const float*)d_in[7];
  const float* ln2b = (const float*)d_in[8];
  const float* w1   = (const float*)d_in[9];
  const float* b1   = (const float*)d_in[10];
  const float* w2   = (const float*)d_in[11];
  const float* b2   = (const float*)d_in[12];
  const float* lnfw = (const float*)d_in[13];
  const float* lnfb = (const float*)d_in[14];
  float* logits = (float*)d_out;

  // workspace carve-up (256B aligned regions)
  char* ws = (char*)d_ws;
  size_t off = 0;
  auto take = [&](size_t bytes) -> char* {
    char* p = ws + off;
    off += (bytes + 255) & ~(size_t)255;
    return p;
  };
  // transposed ([N,K]) f16 weights
  _Float16* tokW  = (_Float16*)take((size_t)Vv * Dv * 2);           // [V,D] already [N,K]
  _Float16* qkvWT = (_Float16*)take((size_t)Lv * Dv * 3 * Dv * 2);  // [L][3D][D]
  _Float16* outWT = (_Float16*)take((size_t)Lv * Dv * Dv * 2);      // [L][D][D]
  _Float16* w1T   = (_Float16*)take((size_t)Lv * Dv * Fv * 2);      // [L][F][D]
  _Float16* w2T   = (_Float16*)take((size_t)Lv * Fv * Dv * 2);      // [L][D][F]
  float*    xbuf  = (float*)   take((size_t)Mrows * Dv * 4);
  _Float16* hbuf  = (_Float16*)take((size_t)Mrows * Dv * 2);
  _Float16* qkvB  = (_Float16*)take((size_t)Mrows * 3 * Dv * 2);
  _Float16* attB  = (_Float16*)take((size_t)Bb * Hh * Tv * Tv * 2);
  _Float16* ybuf  = (_Float16*)take((size_t)Mrows * Dv * 2);
  _Float16* ubuf  = (_Float16*)take((size_t)Mrows * Fv * 2);

  cvt_f32_f16_kernel<<<dim3((unsigned)(((long)Vv * Dv + 255) / 256)), 256, 0, stream>>>(
      tok, tokW, (long)Vv * Dv);
  auto cvtT = [&](const float* src, _Float16* dst, int Kd, int Nd) {
    dim3 g((unsigned)(((long)Kd * Nd + 255) / 256), Lv);
    cvt_transpose_kernel<<<g, 256, 0, stream>>>(src, dst, Kd, Nd);
  };
  cvtT(qkvw, qkvWT, Dv, 3 * Dv);
  cvtT(outw, outWT, Dv, Dv);
  cvtT(w1,   w1T,   Dv, Fv);
  cvtT(w2,   w2T,   Fv, Dv);

  embed_kernel<<<Mrows, 256, 0, stream>>>(ids, tok, pos, xbuf);

  const int D3 = 3 * Dv;                     // 2304
  const long sQKVb = (long)Tv * D3;          // per-batch stride inside qkv buffer

  for (int l = 0; l < Lv; ++l) {
    // LN1 -> h (f16)
    layernorm_f16_kernel<<<Mrows, 256, 0, stream>>>(xbuf, ln1w + l * Dv, ln1b + l * Dv, hbuf);

    // qkv = h @ qkv_w[l]   (B transposed: [3D, D], ldb = D)
    gemm_wmma<true, true, false, false, false><<<dim3(D3 / BNt, Mrows / BMt, 1), 256, 0, stream>>>(
        hbuf, qkvWT + (long)l * Dv * D3, nullptr, qkvB, nullptr, nullptr,
        Mrows, D3, Dv, Dv, Dv, D3, 1, 0, 0, 0, 0, 0, 0);

    // S = Q @ K^T per (b,h): A=Q slice (lda=2304), B=K slice as [N,K] (ldb=2304)
    gemm_wmma<true, true, false, false, false><<<dim3(Tv / BNt, Tv / BMt, Bb * Hh), 256, 0, stream>>>(
        qkvB, qkvB + Dv, nullptr, attB, nullptr, nullptr,
        Tv, Tv, HDv, D3, D3, Tv, Hh,
        sQKVb, (long)HDv, sQKVb, (long)HDv, (long)Hh * Tv * Tv, (long)Tv * Tv);

    // causal softmax with 1/sqrt(64) scale, in place
    softmax_kernel<<<dim3(Tv, Bb * Hh), 256, 0, stream>>>(attB, 0.125f);

    // Y = P @ V per (b,h)  (B = V slice [K=T, N=64], ldb=2304; sync transpose staging)
    gemm_wmma<false, true, false, false, false><<<dim3(HDv / BNt, Tv / BMt, Bb * Hh), 256, 0, stream>>>(
        attB, qkvB + 2 * Dv, nullptr, ybuf, nullptr, nullptr,
        Tv, HDv, Tv, Tv, D3, Dv, Hh,
        (long)Hh * Tv * Tv, (long)Tv * Tv, sQKVb, (long)HDv, (long)Tv * Dv, (long)HDv);

    // x = x + Y @ out_w[l]
    gemm_wmma<true, false, false, false, true><<<dim3(Dv / BNt, Mrows / BMt, 1), 256, 0, stream>>>(
        ybuf, outWT + (long)l * Dv * Dv, xbuf, nullptr, nullptr, xbuf,
        Mrows, Dv, Dv, Dv, Dv, Dv, 1, 0, 0, 0, 0, 0, 0);

    // LN2 -> h
    layernorm_f16_kernel<<<Mrows, 256, 0, stream>>>(xbuf, ln2w + l * Dv, ln2b + l * Dv, hbuf);

    // u = gelu(h @ w1[l] + b1[l])
    gemm_wmma<true, true, true, true, false><<<dim3(Fv / BNt, Mrows / BMt, 1), 256, 0, stream>>>(
        hbuf, w1T + (long)l * Dv * Fv, nullptr, ubuf, b1 + (long)l * Fv, nullptr,
        Mrows, Fv, Dv, Dv, Dv, Fv, 1, 0, 0, 0, 0, 0, 0);

    // x = x + u @ w2[l] + b2[l]
    gemm_wmma<true, false, false, true, true><<<dim3(Dv / BNt, Mrows / BMt, 1), 256, 0, stream>>>(
        ubuf, w2T + (long)l * Fv * Dv, xbuf, nullptr, b2 + (long)l * Dv, xbuf,
        Mrows, Dv, Fv, Fv, Fv, Dv, 1, 0, 0, 0, 0, 0, 0);
  }

  // final LN -> h, then logits = h @ tok_emb^T (tok_emb is [V,D] = [N,K])
  layernorm_f16_kernel<<<Mrows, 256, 0, stream>>>(xbuf, lnfw, lnfb, hbuf);
  gemm_wmma<true, false, false, false, false><<<dim3(Vv / BNt, Mrows / BMt, 1), 256, 0, stream>>>(
      hbuf, tokW, logits, nullptr, nullptr, nullptr,
      Mrows, Vv, Dv, Dv, Dv, Vv, 1, 0, 0, 0, 0, 0, 0);
}
